// DengueGNN_67559835566319
// MI455X (gfx1250) — matrix operand order repaired and checked by reading, not compiled, not run on hardware
//
#include <hip/hip_runtime.h>
#include <hip/hip_bf16.h>

// ---------------------------------------------------------------------------
// CDNA5 (gfx1250) wave32 WMMA helpers: V_WMMA_F32_16X16X32_BF16
// Tile kernels assume N % 16 == 0 (true for this workload: N = 10000).
// ---------------------------------------------------------------------------
typedef __attribute__((ext_vector_type(16))) __bf16 v16bf;
typedef __attribute__((ext_vector_type(8)))  float  v8f;

#define DEVI __device__ __forceinline__

// native fptrunc: backend emits v_cvt_pk_bf16_f32 on gfx1250
DEVI __bf16 f2bf(float x) { return (__bf16)x; }

DEVI v8f vzero8() {
  v8f v;
#pragma unroll
  for (int i = 0; i < 8; ++i) v[i] = 0.0f;
  return v;
}

// order-preserving float<->uint encoding for atomicMax-based segment max
DEVI unsigned enc_f32(float f) {
  unsigned u = __builtin_bit_cast(unsigned, f);
  return (u & 0x80000000u) ? ~u : (u | 0x80000000u);
}
DEVI float dec_f32(unsigned e) {
  unsigned u = (e & 0x80000000u) ? (e & 0x7fffffffu) : ~e;
  return __builtin_bit_cast(float, u);
}
#define ENC_NEG_INF 0x007fffffu   // enc(-inf)

DEVI float sigmf(float x) { return 1.0f / (1.0f + __expf(-x)); }

// ---------------------------------------------------------------------------
// Fragment loaders.
// A-matrix 16x32 bf16 (ISA 7.12.2): lane -> row m=lane&15, half=lane>>4;
// element j -> K = ((j&8)<<1) + half*8 + (j&7). Per lane this is exactly two
// contiguous 8-float runs: [k0+8h, +8) and [k0+16+8h, +8) -> 4x float4 loads.
// ---------------------------------------------------------------------------
template <int LDA, int KA>
DEVI v16bf a_frag_t(const float* A, int row0, int k0, int lane) {
  int m = lane & 15;
  int half = (lane >> 4) & 1;
  const float* rowp = A + (size_t)(row0 + m) * LDA;
  v16bf a;
  if constexpr ((KA & 31) == 0) {
    const float4 u0 = *(const float4*)(rowp + k0 + half * 8);
    const float4 u1 = *(const float4*)(rowp + k0 + half * 8 + 4);
    const float4 u2 = *(const float4*)(rowp + k0 + 16 + half * 8);
    const float4 u3 = *(const float4*)(rowp + k0 + 16 + half * 8 + 4);
    a[0] = f2bf(u0.x);  a[1] = f2bf(u0.y);  a[2] = f2bf(u0.z);  a[3] = f2bf(u0.w);
    a[4] = f2bf(u1.x);  a[5] = f2bf(u1.y);  a[6] = f2bf(u1.z);  a[7] = f2bf(u1.w);
    a[8] = f2bf(u2.x);  a[9] = f2bf(u2.y);  a[10] = f2bf(u2.z); a[11] = f2bf(u2.w);
    a[12] = f2bf(u3.x); a[13] = f2bf(u3.y); a[14] = f2bf(u3.z); a[15] = f2bf(u3.w);
  } else {
    // KA == 16: run0 (K = 8*half .. +7) fully in-bounds; run1 (K>=16) is zero.
    static_assert(KA == 16, "only K=16 tail supported");
    const float4 u0 = *(const float4*)(rowp + half * 8);
    const float4 u1 = *(const float4*)(rowp + half * 8 + 4);
    a[0] = f2bf(u0.x); a[1] = f2bf(u0.y); a[2] = f2bf(u0.z); a[3] = f2bf(u0.w);
    a[4] = f2bf(u1.x); a[5] = f2bf(u1.y); a[6] = f2bf(u1.z); a[7] = f2bf(u1.w);
    const __bf16 z = f2bf(0.0f);
#pragma unroll
    for (int j = 8; j < 16; ++j) a[j] = z;
  }
  return a;
}

// B fragments are pre-swizzled in workspace as [fragIdx][lane][16 bf16]:
// one aligned 32-byte load per lane.
DEVI v16bf b_frag_sw(const __bf16* Bsw, int fragIdx, int lane) {
  return ((const v16bf*)Bsw)[(size_t)fragIdx * 32 + lane];
}

// f32 C/D 16x16: element r -> row = r + 8*half, col = lane%16.
// Full tiles only: 8 unguarded stores at immediate offsets.
template <int LDD>
DEVI void store_d_t(float* D, int row0, int col0, v8f d, int lane) {
  int n = lane & 15, half = (lane >> 4) & 1;
  float* p = D + (size_t)(row0 + 8 * half) * LDD + col0 + n;
#pragma unroll
  for (int r = 0; r < 8; ++r) p[r * LDD] = d[r];
}

DEVI v8f wmma_bf16(v16bf a, v16bf b, v8f c) {
  return __builtin_amdgcn_wmma_f32_16x16x32_bf16(false, a, false, b, (short)0, c,
                                                 false, false);
}

// swizzled element offset for B-fragment storage: logical B[k][c], width NC
DEVI int bswz(int k, int c, int NC) {
  int ks = k >> 5, kk = k & 31;
  int ct = c >> 4, n = c & 15;
  int lane = (((kk >> 3) & 1) << 4) | n;   // half*16 + col
  int j = ((kk >> 4) << 3) | (kk & 7);
  return ((ks * (NC >> 4) + ct) * 32 + lane) * 16 + j;
}

// ---------------------------------------------------------------------------
// Utility / pack kernels
// ---------------------------------------------------------------------------
__global__ void k_fill_f32(float* p, float v, int n) {
  int i = blockIdx.x * blockDim.x + threadIdx.x;
  if (i < n) p[i] = v;
}
__global__ void k_fill_u32(unsigned* p, unsigned v, int n) {
  int i = blockIdx.x * blockDim.x + threadIdx.x;
  if (i < n) p[i] = v;
}
// W0cat logical [32][128]: rows 0-15 = [gat0_W | res0_W]; rows 16-31 = 0
__global__ void k_pack_w0(const float* g0W, const float* r0W, __bf16* dst) {
  int i = blockIdx.x * blockDim.x + threadIdx.x;
  if (i >= 32 * 128) return;
  int k = i >> 7, c = i & 127;
  float v = 0.0f;
  if (k < 16) v = (c < 64) ? g0W[k * 64 + c] : r0W[k * 64 + (c - 64)];
  dst[bswz(k, c, 128)] = f2bf(v);
}
// W1cat logical [64][64]: [gat1_W | res1_W] (each 64x32)
__global__ void k_pack_w1(const float* g1W, const float* r1W, __bf16* dst) {
  int i = blockIdx.x * blockDim.x + threadIdx.x;
  if (i >= 64 * 64) return;
  int k = i >> 6, c = i & 63;
  float v = (c < 32) ? g1W[k * 32 + c] : r1W[k * 32 + (c - 32)];
  dst[bswz(k, c, 64)] = f2bf(v);
}
// logical B[k][c] = src[c*K + k]  (i.e. B = src^T, src is [NC][K] row-major)
__global__ void k_pack_transpose(const float* src, __bf16* dst, int K, int NC) {
  int i = blockIdx.x * blockDim.x + threadIdx.x;
  if (i >= K * NC) return;
  int k = i / NC, c = i - k * NC;
  dst[bswz(k, c, NC)] = f2bf(src[c * K + k]);
}
// logical B[k][c] = src[k*NC + c]
__global__ void k_pack_direct(const float* src, __bf16* dst, int K, int NC) {
  int i = blockIdx.x * blockDim.x + threadIdx.x;
  if (i >= K * NC) return;
  int k = i / NC, c = i - k * NC;
  dst[bswz(k, c, NC)] = f2bf(src[i]);
}

// ---------------------------------------------------------------------------
// Node-tile GEMM: D[N, NC] = A[N, LDA-strided, KA cols](f32) * Bsw(bf16 swz)
// One wave per 16-row tile; fully unrolled; requires N % 16 == 0.
// ---------------------------------------------------------------------------
template <int LDA, int KA, int NC>
__global__ __launch_bounds__(256) void k_gemm_t(const float* __restrict__ A,
                                                const __bf16* __restrict__ Bsw,
                                                float* __restrict__ D, int N) {
  int t = blockIdx.x * blockDim.x + threadIdx.x;
  int wave = t >> 5, lane = t & 31;
  int row0 = wave * 16;
  if (row0 >= N) return;
  constexpr int KSX = (KA + 31) / 32;
  constexpr int NCT = NC / 16;
  v16bf afr[KSX];
#pragma unroll
  for (int ks = 0; ks < KSX; ++ks)
    afr[ks] = a_frag_t<LDA, KA>(A, row0, ks * 32, lane);
#pragma unroll
  for (int ct = 0; ct < NCT; ++ct) {
    v8f acc = vzero8();
#pragma unroll
    for (int ks = 0; ks < KSX; ++ks)
      acc = wmma_bf16(afr[ks], b_frag_sw(Bsw, ks * NCT + ct, lane), acc);
    store_d_t<NC>(D, row0, ct * 16, acc, lane);
  }
}

// al_{s,d}[n*H+h] = sum_c xw[n*ldx + h*32 + c] * a_{src,dst}[h*32+c]
__global__ void k_alpha(const float* xw, int ldx, const float* asrc, const float* adst,
                        float* als, float* ald, int N, int H) {
  int i = blockIdx.x * blockDim.x + threadIdx.x;
  if (i >= N * H) return;
  int n = i / H, h = i - n * H;
  const float* row = xw + (size_t)n * ldx + h * 32;
  float ss = 0.f, sd = 0.f;
#pragma unroll
  for (int c = 0; c < 32; ++c) {
    float v = row[c];
    ss += v * asrc[h * 32 + c];
    sd += v * adst[h * 32 + c];
  }
  als[i] = ss; ald[i] = sd;
}

// per edge: leaky_relu(al_s[src]+al_d[dst]); store score; segment max into mmax[dst]
template <int H>
__global__ void k_edge_score_t(const int* ei, int E, int N, const float* als,
                               const float* ald, float* esc, unsigned* mmax) {
  int e = blockIdx.x * blockDim.x + threadIdx.x;
  int Etot = E + N;
  if (e >= Etot) return;
  int s = (e < E) ? ei[e] : (e - E);
  int d = (e < E) ? ei[E + e] : (e - E);
#pragma unroll
  for (int h = 0; h < H; ++h) {
    float v = als[s * H + h] + ald[d * H + h];
    v = (v > 0.f) ? v : 0.2f * v;
    esc[e * H + h] = v;
    atomicMax(&mmax[d * H + h], enc_f32(v));
  }
}

// wave per edge: ex = exp(e - m[dst]); den[dst] += ex; agg[dst] += ex * xw[src]
template <int H, int LDX>
__global__ __launch_bounds__(256) void k_edge_accum_t(const int* ei, int E, int N,
                                                      const float* esc,
                                                      const unsigned* mmax,
                                                      const float* xw,
                                                      float* den, float* agg) {
  int t = blockIdx.x * blockDim.x + threadIdx.x;
  int e = t >> 5, lane = t & 31;
  int Etot = E + N;
  if (e >= Etot) return;
  int s = (e < E) ? ei[e] : (e - E);
  int d = (e < E) ? ei[E + e] : (e - E);
  if constexpr (H == 2) {
    float ex0 = __expf(esc[e * 2 + 0] - dec_f32(mmax[d * 2 + 0]));
    float ex1 = __expf(esc[e * 2 + 1] - dec_f32(mmax[d * 2 + 1]));
    if (lane == 0) atomicAdd(&den[d * 2 + 0], ex0);
    if (lane == 1) atomicAdd(&den[d * 2 + 1], ex1);
    atomicAdd(&agg[(size_t)d * 64 + lane],      ex0 * xw[(size_t)s * LDX + lane]);
    atomicAdd(&agg[(size_t)d * 64 + 32 + lane], ex1 * xw[(size_t)s * LDX + 32 + lane]);
  } else {
    float ex = __expf(esc[e] - dec_f32(mmax[d]));
    if (lane == 0) atomicAdd(&den[d], ex);
    atomicAdd(&agg[(size_t)d * 32 + lane], ex * xw[(size_t)s * LDX + lane]);
  }
}

// x_out = elu(agg/(den+1e-16) + b) + res   (res = xwr[:, roff:roff+W])
__global__ void k_gat_finalize(const float* agg, const float* den, const float* b,
                               const float* xwr, int ldr, int roff,
                               float* xout, int N, int W, int H) {
  int i = blockIdx.x * blockDim.x + threadIdx.x;
  if (i >= N * W) return;
  int n = i / W, c = i - n * W;
  int h = (H == 2) ? (c >> 5) : 0;
  float v = agg[i] / (den[n * H + h] + 1e-16f) + b[c];
  v = (v > 0.f) ? v : (__expf(v) - 1.0f);
  xout[i] = v + xwr[(size_t)n * ldr + roff + c];
}

// ---------------------------------------------------------------------------
// Fused GRU cell: h <- GRU(x, h). One wave per 16-node tile; N % 16 == 0.
// In-place safe: A-fragments of h preloaded before any write; each column tile
// touches only its own columns for the elementwise update.
// ---------------------------------------------------------------------------
template <int KX>
__global__ __launch_bounds__(256) void k_gru_t(const float* __restrict__ x,
                                               float* __restrict__ h,
                                               const __bf16* __restrict__ WiSw,
                                               const __bf16* __restrict__ WhSw,
                                               const float* bi, const float* bh,
                                               int N) {
  int t = blockIdx.x * blockDim.x + threadIdx.x;
  int wave = t >> 5, lane = t & 31;
  int row0 = wave * 16;
  if (row0 >= N) return;
  int half = (lane >> 4) & 1, nlo = lane & 15;
  constexpr int KSX = KX / 32;   // 1 or 2

  v16bf ax[KSX], ah[2];
#pragma unroll
  for (int ks = 0; ks < KSX; ++ks)
    ax[ks] = a_frag_t<KX, KX>(x, row0, ks * 32, lane);
  ah[0] = a_frag_t<64, 64>(h, row0, 0, lane);
  ah[1] = a_frag_t<64, 64>(h, row0, 32, lane);

#pragma unroll
  for (int ct = 0; ct < 4; ++ct) {
    v8f gir = vzero8(), giz = vzero8(), gin = vzero8();
    v8f ghr = vzero8(), ghz = vzero8(), ghn = vzero8();
#pragma unroll
    for (int ks = 0; ks < KSX; ++ks) {          // Wi^T is [KX][192] -> 12 col tiles
      gir = wmma_bf16(ax[ks], b_frag_sw(WiSw, ks * 12 + ct, lane), gir);
      giz = wmma_bf16(ax[ks], b_frag_sw(WiSw, ks * 12 + 4 + ct, lane), giz);
      gin = wmma_bf16(ax[ks], b_frag_sw(WiSw, ks * 12 + 8 + ct, lane), gin);
    }
#pragma unroll
    for (int ks = 0; ks < 2; ++ks) {            // Wh^T is [64][192]
      ghr = wmma_bf16(ah[ks], b_frag_sw(WhSw, ks * 12 + ct, lane), ghr);
      ghz = wmma_bf16(ah[ks], b_frag_sw(WhSw, ks * 12 + 4 + ct, lane), ghz);
      ghn = wmma_bf16(ah[ks], b_frag_sw(WhSw, ks * 12 + 8 + ct, lane), ghn);
    }
    int c = ct * 16 + nlo;
    float bir = bi[c], biz = bi[64 + c], bin = bi[128 + c];
    float bhr = bh[c], bhz = bh[64 + c], bhn = bh[128 + c];
    float* hp = h + (size_t)(row0 + 8 * half) * 64 + c;
#pragma unroll
    for (int r = 0; r < 8; ++r) {
      float hold = hp[r * 64];
      float rg = sigmf(gir[r] + bir + ghr[r] + bhr);
      float zg = sigmf(giz[r] + biz + ghz[r] + bhz);
      float ng = tanhf(gin[r] + bin + rg * (ghn[r] + bhn));
      hp[r * 64] = (1.0f - zg) * ng + zg * hold;
    }
  }
}

// out[n] = fc2_b + sum_c relu(hf[n,c] + fc1_b[c]) * fc2_W[c]
__global__ void k_head(const float* hf, const float* b1, const float* w2,
                       const float* b2, float* out, int N) {
  int n = blockIdx.x * blockDim.x + threadIdx.x;
  if (n >= N) return;
  float acc = b2[0];
#pragma unroll
  for (int c = 0; c < 32; ++c) {
    float v = hf[(size_t)n * 32 + c] + b1[c];
    v = (v > 0.f) ? v : 0.f;
    acc += v * w2[c];
  }
  out[n] = acc;
}

// ---------------------------------------------------------------------------
extern "C" void kernel_launch(void* const* d_in, const int* in_sizes, int n_in,
                              void* d_out, int out_size, void* d_ws, size_t ws_size,
                              hipStream_t stream) {
  (void)n_in; (void)out_size; (void)ws_size;
  const float* x_seq = (const float*)d_in[0];
  const int*   ei    = (const int*)d_in[1];
  const float* g0W   = (const float*)d_in[2];
  const float* g0as  = (const float*)d_in[3];
  const float* g0ad  = (const float*)d_in[4];
  const float* g0b   = (const float*)d_in[5];
  const float* r0W   = (const float*)d_in[6];
  const float* g1W   = (const float*)d_in[7];
  const float* g1as  = (const float*)d_in[8];
  const float* g1ad  = (const float*)d_in[9];
  const float* g1b   = (const float*)d_in[10];
  const float* r1W   = (const float*)d_in[11];
  const float* g0Wi  = (const float*)d_in[12];
  const float* g0Wh  = (const float*)d_in[13];
  const float* g0bi  = (const float*)d_in[14];
  const float* g0bh  = (const float*)d_in[15];
  const float* g1Wi  = (const float*)d_in[16];
  const float* g1Wh  = (const float*)d_in[17];
  const float* g1bi  = (const float*)d_in[18];
  const float* g1bh  = (const float*)d_in[19];
  const float* fc1W  = (const float*)d_in[20];
  const float* fc1b  = (const float*)d_in[21];
  const float* fc2W  = (const float*)d_in[22];
  const float* fc2b  = (const float*)d_in[23];

  const int T = 5, FIN = 16;
  const int E = in_sizes[1] / 2;
  const int N = in_sizes[0] / (T * FIN);   // N = 10000, multiple of 16
  const int Etot = E + N;
  const int tiles = N / 16;                // full tiles only

  size_t off = 0;
  auto wsa = [&](size_t bytes) -> void* {
    off = (off + 255) & ~(size_t)255;
    void* p = (char*)d_ws + off;
    off += bytes;
    return p;
  };
  __bf16* W0cat = (__bf16*)wsa((size_t)32 * 128 * 2);
  __bf16* W1cat = (__bf16*)wsa((size_t)64 * 64 * 2);
  __bf16* G0WiT = (__bf16*)wsa((size_t)32 * 192 * 2);
  __bf16* G0WhT = (__bf16*)wsa((size_t)64 * 192 * 2);
  __bf16* G1WiT = (__bf16*)wsa((size_t)64 * 192 * 2);
  __bf16* G1WhT = (__bf16*)wsa((size_t)64 * 192 * 2);
  __bf16* FC1bf = (__bf16*)wsa((size_t)64 * 32 * 2);
  float* xwr0 = (float*)wsa((size_t)N * 128 * 4);
  float* al0s = (float*)wsa((size_t)N * 2 * 4);
  float* al0d = (float*)wsa((size_t)N * 2 * 4);
  float* esc0 = (float*)wsa((size_t)Etot * 2 * 4);
  unsigned* mmax0 = (unsigned*)wsa((size_t)N * 2 * 4);
  float* den0 = (float*)wsa((size_t)N * 2 * 4);
  float* agg0 = (float*)wsa((size_t)N * 64 * 4);
  float* x1   = (float*)wsa((size_t)N * 64 * 4);
  float* xwr1 = (float*)wsa((size_t)N * 64 * 4);
  float* al1s = (float*)wsa((size_t)N * 4);
  float* al1d = (float*)wsa((size_t)N * 4);
  float* esc1 = (float*)wsa((size_t)Etot * 4);
  unsigned* mmax1 = (unsigned*)wsa((size_t)N * 4);
  float* den1 = (float*)wsa((size_t)N * 4);
  float* agg1 = (float*)wsa((size_t)N * 32 * 4);
  float* x2   = (float*)wsa((size_t)N * 32 * 4);
  float* h0   = (float*)wsa((size_t)N * 64 * 4);
  float* h1   = (float*)wsa((size_t)N * 64 * 4);
  float* hfc  = (float*)wsa((size_t)N * 32 * 4);

  const int B = 256;
  auto gr = [](int n) { return (n + 255) / 256; };
  const int gemmB = (tiles * 32 + B - 1) / B;

  // ---- one-time weight prep (cheap, deterministic every call) ----
  k_pack_w0<<<gr(32 * 128), B, 0, stream>>>(g0W, r0W, W0cat);
  k_pack_w1<<<gr(64 * 64), B, 0, stream>>>(g1W, r1W, W1cat);
  k_pack_transpose<<<gr(32 * 192), B, 0, stream>>>(g0Wi, G0WiT, 32, 192);
  k_pack_transpose<<<gr(64 * 192), B, 0, stream>>>(g0Wh, G0WhT, 64, 192);
  k_pack_transpose<<<gr(64 * 192), B, 0, stream>>>(g1Wi, G1WiT, 64, 192);
  k_pack_transpose<<<gr(64 * 192), B, 0, stream>>>(g1Wh, G1WhT, 64, 192);
  k_pack_direct<<<gr(64 * 32), B, 0, stream>>>(fc1W, FC1bf, 64, 32);
  k_fill_f32<<<gr(N * 64), B, 0, stream>>>(h0, 0.f, N * 64);
  k_fill_f32<<<gr(N * 64), B, 0, stream>>>(h1, 0.f, N * 64);

  for (int t = 0; t < T; ++t) {
    const float* xt = x_seq + (size_t)t * N * FIN;
    // reset per-step accumulators
    k_fill_u32<<<gr(N * 2), B, 0, stream>>>(mmax0, ENC_NEG_INF, N * 2);
    k_fill_f32<<<gr(N * 2), B, 0, stream>>>(den0, 0.f, N * 2);
    k_fill_f32<<<gr(N * 64), B, 0, stream>>>(agg0, 0.f, N * 64);
    k_fill_u32<<<gr(N), B, 0, stream>>>(mmax1, ENC_NEG_INF, N);
    k_fill_f32<<<gr(N), B, 0, stream>>>(den1, 0.f, N);
    k_fill_f32<<<gr(N * 32), B, 0, stream>>>(agg1, 0.f, N * 32);

    // GAT layer 0: [x|res] transform 16 -> 128 cols
    k_gemm_t<16, 16, 128><<<gemmB, B, 0, stream>>>(xt, W0cat, xwr0, N);
    k_alpha<<<gr(N * 2), B, 0, stream>>>(xwr0, 128, g0as, g0ad, al0s, al0d, N, 2);
    k_edge_score_t<2><<<gr(Etot), B, 0, stream>>>(ei, E, N, al0s, al0d, esc0, mmax0);
    k_edge_accum_t<2, 128><<<(Etot * 32 + B - 1) / B, B, 0, stream>>>(
        ei, E, N, esc0, mmax0, xwr0, den0, agg0);
    k_gat_finalize<<<gr(N * 64), B, 0, stream>>>(agg0, den0, g0b, xwr0, 128, 64,
                                                 x1, N, 64, 2);
    // GAT layer 1: 64 -> 64 cols ([xw|res])
    k_gemm_t<64, 64, 64><<<gemmB, B, 0, stream>>>(x1, W1cat, xwr1, N);
    k_alpha<<<gr(N), B, 0, stream>>>(xwr1, 64, g1as, g1ad, al1s, al1d, N, 1);
    k_edge_score_t<1><<<gr(Etot), B, 0, stream>>>(ei, E, N, al1s, al1d, esc1, mmax1);
    k_edge_accum_t<1, 64><<<(Etot * 32 + B - 1) / B, B, 0, stream>>>(
        ei, E, N, esc1, mmax1, xwr1, den1, agg1);
    k_gat_finalize<<<gr(N * 32), B, 0, stream>>>(agg1, den1, g1b, xwr1, 64, 32,
                                                 x2, N, 32, 1);
    // GRU stack
    k_gru_t<32><<<gemmB, B, 0, stream>>>(x2, h0, G0WiT, G0WhT, g0bi, g0bh, N);
    k_gru_t<64><<<gemmB, B, 0, stream>>>(h0, h1, G1WiT, G1WhT, g1bi, g1bh, N);
  }

  // MLP head
  k_gemm_t<64, 64, 32><<<gemmB, B, 0, stream>>>(h1, FC1bf, hfc, N);
  k_head<<<gr(N), B, 0, stream>>>(hfc, fc1b, fc2W, fc2b, (float*)d_out, N);
}
